// DGCRM_56075093017380
// MI455X (gfx1250) — matrix operand order repaired and checked by reading, not compiled, not run on hardware
//
#include <hip/hip_runtime.h>
#include <hip/hip_bf16.h>
#include <math.h>

#define B_   16
#define T_   12
#define N_   8600
#define DIN_ 2
#define H_   64
#define E2_  128     // 2*H
#define KIN_ 66      // DIN + H
#define KP_  96      // K padded to 3 * 32 for wmma_f32_16x16x32_f16

typedef _Float16 v16h __attribute__((ext_vector_type(16)));
typedef _Float16 v8h  __attribute__((ext_vector_type(8)));
typedef float    v8f  __attribute__((ext_vector_type(8)));

union V16U { v16h v; v8h h[2]; };

template<bool V> struct BoolC { static constexpr bool value = V; };

// A-fragment (16x32 f16): lane m=lane%16 holds row m; half=lane/16 selects K-phase.
__device__ __forceinline__ v16h load_afrag(const _Float16* row, int kbase, int half) {
    V16U u;
    u.h[0] = *(const v8h*)(row + kbase + 8 * half);
    u.h[1] = *(const v8h*)(row + kbase + 16 + 8 * half);
    return u.v;
}

// B-fragment (32x16 f16): lane holds column e, K = kbase + 8*half + idx (idx<8), kbase + 16 + 8*half + idx-8 (idx>=8).
// Weights stored transposed in LDS as wt[e*KP_ + k] so this is 2x ds_load_b128.
__device__ __forceinline__ v16h load_bfrag(const _Float16* wt, int e, int kbase, int half) {
    const _Float16* p = wt + e * KP_ + kbase + 8 * half;
    V16U u;
    u.h[0] = *(const v8h*)(p);
    u.h[1] = *(const v8h*)(p + 16);
    return u.v;
}

__device__ __forceinline__ v8f wmma16(v16h a, v16h b, v8f c) {
    return __builtin_amdgcn_wmma_f32_16x16x32_f16(false, a, false, b, (short)0, c, false, false);
}

__device__ __forceinline__ float sigmoidf_(float x) { return 1.0f / (1.0f + __expf(-x)); }

__global__ void kZero(float* p, int n) {
    int i = blockIdx.x * blockDim.x + threadIdx.x;
    if (i < n) p[i] = 0.0f;
}

// ---------------- Kernel A: gate h = relu(inp @ gate_w + b), column-sum -> aggG ----
__global__ __launch_bounds__(256) void kA(const float* __restrict__ x,
                                          const float* __restrict__ prevState, size_t pstr,
                                          const float* __restrict__ gate_w,
                                          const float* __restrict__ gate_b,
                                          float* __restrict__ aggG, int t) {
    __shared__ alignas(16) _Float16 rows[128 * KP_];
    __shared__ alignas(16) _Float16 wt[E2_ * KP_];
    __shared__ float red[E2_];
    const int b = blockIdx.y;
    const int n0 = blockIdx.x * 128;
    const int tid = threadIdx.x;
    const bool fullTile = (n0 + 128 <= N_);

    if (tid < E2_) red[tid] = 0.0f;

    const float* xrow = x + ((size_t)(b * T_ + t)) * N_ * DIN_;

    auto body = [&](auto full_c) {
        constexpr bool FULL = decltype(full_c)::value;
        for (int i = tid; i < 128 * KP_; i += 256) {
            int ridx = i / KP_;
            int e = i - ridx * KP_;
            int n = n0 + ridx;
            float v = 0.0f;
            if (FULL || n < N_) {
                if (e < DIN_)      v = xrow[(size_t)n * DIN_ + e];
                else if (e < KIN_) v = prevState[((size_t)b * pstr + n) * H_ + (e - DIN_)];
            }
            rows[i] = (_Float16)v;
        }
        for (int i = tid; i < E2_ * KP_; i += 256) {
            int e = i / KP_;
            int k = i - e * KP_;
            wt[i] = (_Float16)((k < KIN_) ? gate_w[(size_t)k * E2_ + e] : 0.0f);
        }
        __syncthreads();

        const int wave = tid >> 5, lane = tid & 31, m = lane & 15, half = lane >> 4;
        const _Float16* arow = rows + (wave * 16 + m) * KP_;
        v16h a0 = load_afrag(arow, 0, half);
        v16h a1 = load_afrag(arow, 32, half);
        v16h a2 = load_afrag(arow, 64, half);

#pragma unroll
        for (int j = 0; j < 8; ++j) {
            int e = j * 16 + m;
            v8f acc = {};
            acc = wmma16(a0, load_bfrag(wt, e, 0,  half), acc);
            acc = wmma16(a1, load_bfrag(wt, e, 32, half), acc);
            acc = wmma16(a2, load_bfrag(wt, e, 64, half), acc);
            float bias = gate_b[e];
            float psum = 0.0f;
#pragma unroll
            for (int r = 0; r < 8; ++r) {
                float v = acc[r] + bias;
                v = v > 0.0f ? v : 0.0f;
                if (FULL) {
                    psum += v;
                } else {
                    int n = n0 + wave * 16 + r + 8 * half;
                    if (n < N_) psum += v;
                }
            }
            atomicAdd(&red[e], psum);   // ds_add_f32
        }
    };
    if (fullTile) body(BoolC<true>{}); else body(BoolC<false>{});

    __syncthreads();
    if (tid < E2_) unsafeAtomicAdd(&aggG[(t * B_ + b) * E2_ + tid], red[tid]);
}

// ---------------- Kernel B: res_g, zr=sigmoid, z*state & r out, cand matmul -> aggU --
__global__ __launch_bounds__(256) void kB(const float* __restrict__ x,
                                          const float* __restrict__ prevState, size_t pstr,
                                          const float* __restrict__ gate_align_w,
                                          const float* __restrict__ gate_align_b,
                                          const float* __restrict__ gate_node_w,
                                          const float* __restrict__ gate_add_w,
                                          const float* __restrict__ gate_aff_w,
                                          const float* __restrict__ gate_aff_b,
                                          const float* __restrict__ upd_w,
                                          const float* __restrict__ upd_b,
                                          const float* __restrict__ aggG,
                                          float* __restrict__ aggU,
                                          float* __restrict__ rbuf,
                                          float* __restrict__ zsbuf, int t) {
    __shared__ alignas(16) _Float16 rows[128 * KP_];
    __shared__ alignas(16) _Float16 wtA[E2_ * KP_];
    __shared__ alignas(16) _Float16 wtU[H_ * KP_];
    __shared__ float red[H_];
    const int b = blockIdx.y;
    const int n0 = blockIdx.x * 128;
    const int tid = threadIdx.x;
    const bool fullTile = (n0 + 128 <= N_);

    if (tid < H_) red[tid] = 0.0f;

    const float* xrow = x + ((size_t)(b * T_ + t)) * N_ * DIN_;

    auto body = [&](auto full_c) {
        constexpr bool FULL = decltype(full_c)::value;
        for (int i = tid; i < 128 * KP_; i += 256) {
            int ridx = i / KP_;
            int e = i - ridx * KP_;
            int n = n0 + ridx;
            float v = 0.0f;
            if (FULL || n < N_) {
                if (e < DIN_)      v = xrow[(size_t)n * DIN_ + e];
                else if (e < KIN_) v = prevState[((size_t)b * pstr + n) * H_ + (e - DIN_)];
            }
            rows[i] = (_Float16)v;
        }
        for (int i = tid; i < E2_ * KP_; i += 256) {
            int e = i / KP_;
            int k = i - e * KP_;
            wtA[i] = (_Float16)((k < KIN_) ? gate_align_w[(size_t)k * E2_ + e] : 0.0f);
        }
        for (int i = tid; i < H_ * KP_; i += 256) {
            int e = i / KP_;
            int k = i - e * KP_;
            wtU[i] = (_Float16)((k < KIN_) ? upd_w[(size_t)k * H_ + e] : 0.0f);
        }
        __syncthreads();

        const int wave = tid >> 5, lane = tid & 31, m = lane & 15, half = lane >> 4;
        _Float16* arow = rows + (wave * 16 + m) * KP_;
        v16h a0 = load_afrag(arow, 0, half);
        v16h a1 = load_afrag(arow, 32, half);
        v16h a2 = load_afrag(arow, 64, half);

        v8f acc[8];
#pragma unroll
        for (int j = 0; j < 8; ++j) {
            int e = j * 16 + m;
            v8f c = {};
            c = wmma16(a0, load_bfrag(wtA, e, 0,  half), c);
            c = wmma16(a1, load_bfrag(wtA, e, 32, half), c);
            c = wmma16(a2, load_bfrag(wtA, e, 64, half), c);
            acc[j] = c;
        }

        float aG[8], bias[8];
#pragma unroll
        for (int j = 0; j < 8; ++j) {
            int e = j * 16 + m;
            aG[j]   = aggG[(t * B_ + b) * E2_ + e];
            bias[j] = gate_align_b[e];
        }

        // elementwise: zr = sigmoid(res + aff_w * (s * aggG) + aff_b); split z / r
#pragma unroll
        for (int r = 0; r < 8; ++r) {
            int row = wave * 16 + r + 8 * half;
            int n = n0 + row;
            if (FULL || n < N_) {
                float sg = gate_add_w[n] * gate_node_w[n];
                const float* affw  = gate_aff_w + (size_t)n * E2_;
                const float* affb  = gate_aff_b + (size_t)n * E2_;
                const float* strow = prevState + ((size_t)b * pstr + n) * H_;
                float* zsrow = zsbuf + ((size_t)b * N_ + n) * H_;
                float* rrow  = rbuf  + ((size_t)b * N_ + n) * H_;
#pragma unroll
                for (int j = 0; j < 8; ++j) {
                    int e = j * 16 + m;
                    float val = acc[j][r] + bias[j] + affw[e] * (sg * aG[j]) + affb[e];
                    float zr = sigmoidf_(val);
                    if (j < 4) {             // z-half: h = e
                        float zs = zr * strow[e];
                        zsrow[e] = zs;
                        rows[row * KP_ + DIN_ + e] = (_Float16)zs;  // cand row in place
                    } else {                 // r-half: h = e - H
                        rrow[e - H_] = zr;
                    }
                }
            }
        }
        __syncthreads();

        // cand @ upd_w, relu, column-sum -> aggU
        v16h c0 = load_afrag(arow, 0, half);
        v16h c1 = load_afrag(arow, 32, half);
        v16h c2 = load_afrag(arow, 64, half);
#pragma unroll
        for (int j = 0; j < 4; ++j) {
            int e = j * 16 + m;
            v8f c = {};
            c = wmma16(c0, load_bfrag(wtU, e, 0,  half), c);
            c = wmma16(c1, load_bfrag(wtU, e, 32, half), c);
            c = wmma16(c2, load_bfrag(wtU, e, 64, half), c);
            float ub = upd_b[e];
            float psum = 0.0f;
#pragma unroll
            for (int r = 0; r < 8; ++r) {
                float v = c[r] + ub;
                v = v > 0.0f ? v : 0.0f;
                if (FULL) {
                    psum += v;
                } else {
                    int n = n0 + wave * 16 + r + 8 * half;
                    if (n < N_) psum += v;
                }
            }
            atomicAdd(&red[e], psum);
        }
    };
    if (fullTile) body(BoolC<true>{}); else body(BoolC<false>{});

    __syncthreads();
    if (tid < H_) unsafeAtomicAdd(&aggU[(t * B_ + b) * H_ + tid], red[tid]);
}

// ---------------- Kernel C: hc = tanh(res_u + ...), h = r*s + (1-r)*hc, write out ----
__global__ __launch_bounds__(256) void kC(const float* __restrict__ x,
                                          const float* __restrict__ prevState, size_t pstr,
                                          const float* __restrict__ upd_align_w,
                                          const float* __restrict__ upd_align_b,
                                          const float* __restrict__ upd_node_w,
                                          const float* __restrict__ upd_add_w,
                                          const float* __restrict__ upd_aff_w,
                                          const float* __restrict__ upd_aff_b,
                                          const float* __restrict__ aggU,
                                          const float* __restrict__ rbuf,
                                          const float* __restrict__ zsbuf,
                                          float* __restrict__ outStates,
                                          float* __restrict__ outLast, int t) {
    __shared__ alignas(16) _Float16 rows[128 * KP_];
    __shared__ alignas(16) _Float16 wtUA[H_ * KP_];
    const int b = blockIdx.y;
    const int n0 = blockIdx.x * 128;
    const int tid = threadIdx.x;
    const bool fullTile = (n0 + 128 <= N_);

    const float* xrow = x + ((size_t)(b * T_ + t)) * N_ * DIN_;
    float* outS = outStates + ((size_t)b * T_ + t) * N_ * H_;

    auto body = [&](auto full_c) {
        constexpr bool FULL = decltype(full_c)::value;
        for (int i = tid; i < 128 * KP_; i += 256) {
            int ridx = i / KP_;
            int e = i - ridx * KP_;
            int n = n0 + ridx;
            float v = 0.0f;
            if (FULL || n < N_) {
                if (e < DIN_)      v = xrow[(size_t)n * DIN_ + e];
                else if (e < KIN_) v = zsbuf[((size_t)b * N_ + n) * H_ + (e - DIN_)];
            }
            rows[i] = (_Float16)v;
        }
        for (int i = tid; i < H_ * KP_; i += 256) {
            int e = i / KP_;
            int k = i - e * KP_;
            wtUA[i] = (_Float16)((k < KIN_) ? upd_align_w[(size_t)k * H_ + e] : 0.0f);
        }
        __syncthreads();

        const int wave = tid >> 5, lane = tid & 31, m = lane & 15, half = lane >> 4;
        const _Float16* arow = rows + (wave * 16 + m) * KP_;
        v16h a0 = load_afrag(arow, 0, half);
        v16h a1 = load_afrag(arow, 32, half);
        v16h a2 = load_afrag(arow, 64, half);

        v8f acc[4];
        float aU[4], bias[4];
#pragma unroll
        for (int j = 0; j < 4; ++j) {
            int e = j * 16 + m;
            v8f c = {};
            c = wmma16(a0, load_bfrag(wtUA, e, 0,  half), c);
            c = wmma16(a1, load_bfrag(wtUA, e, 32, half), c);
            c = wmma16(a2, load_bfrag(wtUA, e, 64, half), c);
            acc[j]  = c;
            aU[j]   = aggU[(t * B_ + b) * H_ + e];
            bias[j] = upd_align_b[e];
        }

#pragma unroll
        for (int r = 0; r < 8; ++r) {
            int n = n0 + wave * 16 + r + 8 * half;
            if (FULL || n < N_) {
                float su = upd_add_w[n] * upd_node_w[n];
                const float* affw  = upd_aff_w + (size_t)n * H_;
                const float* affb  = upd_aff_b + (size_t)n * H_;
                const float* rrow  = rbuf + ((size_t)b * N_ + n) * H_;
                const float* strow = prevState + ((size_t)b * pstr + n) * H_;
                float* orow = outS + (size_t)n * H_;
                float* lrow = outLast + ((size_t)b * N_ + n) * H_;
#pragma unroll
                for (int j = 0; j < 4; ++j) {
                    int e = j * 16 + m;
                    float val = acc[j][r] + bias[j] + affw[e] * (su * aU[j]) + affb[e];
                    float hc = tanhf(val);
                    float rv = rrow[e];
                    float st = strow[e];
                    float hn = rv * st + (1.0f - rv) * hc;
                    orow[e] = hn;
                    if (t == T_ - 1) lrow[e] = hn;
                }
            }
        }
    };
    if (fullTile) body(BoolC<true>{}); else body(BoolC<false>{});
}

extern "C" void kernel_launch(void* const* d_in, const int* in_sizes, int n_in,
                              void* d_out, int out_size, void* d_ws, size_t ws_size,
                              hipStream_t stream) {
    const float* x            = (const float*)d_in[0];
    const float* init_state   = (const float*)d_in[1];
    const float* gate_align_w = (const float*)d_in[4];
    const float* gate_align_b = (const float*)d_in[5];
    const float* gate_w       = (const float*)d_in[6];
    const float* gate_b       = (const float*)d_in[7];
    const float* gate_node_w  = (const float*)d_in[8];
    const float* gate_add_w   = (const float*)d_in[9];
    const float* gate_aff_w   = (const float*)d_in[10];
    const float* gate_aff_b   = (const float*)d_in[11];
    const float* upd_align_w  = (const float*)d_in[12];
    const float* upd_align_b  = (const float*)d_in[13];
    const float* upd_w        = (const float*)d_in[14];
    const float* upd_b        = (const float*)d_in[15];
    const float* upd_node_w   = (const float*)d_in[16];
    const float* upd_add_w    = (const float*)d_in[17];
    const float* upd_aff_w    = (const float*)d_in[18];
    const float* upd_aff_b    = (const float*)d_in[19];

    float* outStates = (float*)d_out;
    float* outLast   = outStates + (size_t)B_ * T_ * N_ * H_;

    float* ws    = (float*)d_ws;
    float* aggG  = ws;                                   // T*B*128
    float* aggU  = aggG + T_ * B_ * E2_;                 // T*B*64
    float* rbuf  = aggU + T_ * B_ * H_;                  // B*N*H
    float* zsbuf = rbuf + (size_t)B_ * N_ * H_;          // B*N*H

    const int nz = T_ * B_ * E2_ + T_ * B_ * H_;
    kZero<<<(nz + 255) / 256, 256, 0, stream>>>(aggG, nz);

    dim3 grd((N_ + 127) / 128, B_);

    for (int t = 0; t < T_; ++t) {
        const float* prev;
        size_t pstr;
        if (t == 0) { prev = init_state; pstr = (size_t)N_; }
        else        { prev = outStates + (size_t)(t - 1) * N_ * H_; pstr = (size_t)T_ * N_; }

        kA<<<grd, 256, 0, stream>>>(x, prev, pstr, gate_w, gate_b, aggG, t);
        kB<<<grd, 256, 0, stream>>>(x, prev, pstr, gate_align_w, gate_align_b,
                                    gate_node_w, gate_add_w, gate_aff_w, gate_aff_b,
                                    upd_w, upd_b, aggG, aggU, rbuf, zsbuf, t);
        kC<<<grd, 256, 0, stream>>>(x, prev, pstr, upd_align_w, upd_align_b,
                                    upd_node_w, upd_add_w, upd_aff_w, upd_aff_b,
                                    aggU, rbuf, zsbuf, outStates, outLast, t);
    }
}
